// APGAttention_83751862272086
// MI455X (gfx1250) — matrix operand order repaired and checked by reading, not compiled, not run on hardware
//
#include <hip/hip_runtime.h>
#include <hip/hip_bf16.h>

#define DIM   768
#define SEQ   2048
#define BATCH 4
#define NTOK  (BATCH * SEQ)   // 8192

typedef __attribute__((ext_vector_type(16))) _Float16 v16h;
typedef __attribute__((ext_vector_type(8)))  _Float16 v8h;
typedef __attribute__((ext_vector_type(8)))  float    v8f;
typedef __attribute__((ext_vector_type(4)))  float    v4f;

static __device__ inline v8f wmma16(v16h a, v16h b, v8f c) {
  // D = A(16x32 f16) x B(32x16 f16) + C(16x16 f32)
  return __builtin_amdgcn_wmma_f32_16x16x32_f16(false, a, false, b, (short)0, c, false, false);
}

static __device__ inline v8f zero8() {
  v8f z = {0.f, 0.f, 0.f, 0.f, 0.f, 0.f, 0.f, 0.f};
  return z;
}

// ---- fragment loaders (all-f16) --------------------------------------------
// A-fragment 16x32 (MxK) from row-major [*, ld]:
//   lane (m = lane&15, hi = lane>>4): h[0..7] = X[m][k0+8*hi..+7], h[8..15] = X[m][k0+16+8*hi..+7]
static __device__ inline v16h load_a16(const _Float16* base, int row0, int k0, int ld) {
  const int lane = threadIdx.x & 31;
  const int m = lane & 15, hi = lane >> 4;
  const _Float16* p = base + (size_t)(row0 + m) * ld + k0 + 8 * hi;
  v8h lo = *(const v8h*)p;
  v8h hh = *(const v8h*)(p + 16);
  return __builtin_shufflevector(lo, hh, 0,1,2,3,4,5,6,7,8,9,10,11,12,13,14,15);
}

// B-fragment 32x16 (KxN) where column n of B is contiguous in memory:
//   B[k][n] = base[(n0+n)*ld + k0 + k];  lane holds h[0..7] = B[8*hi..+7][lane&15], h[8..15] = B[16+8*hi..+7][lane&15]
static __device__ inline v16h load_b16(const _Float16* base, int n0, int k0, int ld) {
  const int lane = threadIdx.x & 31;
  const int n = lane & 15, hi = lane >> 4;
  const _Float16* p = base + (size_t)(n0 + n) * ld + k0 + 8 * hi;
  v8h lo = *(const v8h*)p;
  v8h hh = *(const v8h*)(p + 16);
  return __builtin_shufflevector(lo, hh, 0,1,2,3,4,5,6,7,8,9,10,11,12,13,14,15);
}

// ---- kernel 0: fp32 -> f16 bulk convert ------------------------------------
__global__ __launch_bounds__(256) void cvt_kernel(const float* __restrict__ src,
                                                  _Float16* __restrict__ dst, int n) {
  const int i = (blockIdx.x * 256 + threadIdx.x) * 8;
  if (i + 8 <= n) {
    v4f a = *(const v4f*)(src + i);
    v4f b = *(const v4f*)(src + i + 4);
    v8h r;
    #pragma unroll
    for (int j = 0; j < 4; ++j) { r[j] = (_Float16)a[j]; r[j + 4] = (_Float16)b[j]; }
    *(v8h*)(dst + i) = r;
  }
}

// ---- kernel 1: q/k/v projections (blockIdx.y selects which) -----------------
// out[r][c] = sum_d X[r][d]*W[c][d] + b[c]; q gets * DIM^-0.5; v stored transposed.
// Manually double-buffered k-loop so fragment loads overlap WMMA execution.
__global__ __launch_bounds__(256) void qkv_proj_kernel(
    const _Float16* __restrict__ qh, const _Float16* __restrict__ kh,
    const _Float16* __restrict__ vh,
    const _Float16* __restrict__ Wqh, const _Float16* __restrict__ Wkh,
    const _Float16* __restrict__ Wvh,
    const float* __restrict__ bq, const float* __restrict__ bk,
    const float* __restrict__ bv,
    _Float16* __restrict__ qp, _Float16* __restrict__ kp, _Float16* __restrict__ vpT)
{
  const int which = blockIdx.y;
  const _Float16* X    = (which == 0) ? qh  : (which == 1) ? kh  : vh;
  const _Float16* W    = (which == 0) ? Wqh : (which == 1) ? Wkh : Wvh;
  const float*    bias = (which == 0) ? bq  : (which == 1) ? bk  : bv;
  const int r0 = blockIdx.x * 16;
  const int wave = threadIdx.x >> 5;
  const int lane = threadIdx.x & 31;
  const int hi = lane >> 4, ln = lane & 15;
  const int c0 = wave * 96;                   // 8 waves * 96 = 768 columns
  const float scale = 0.03608439182435161f;   // 768^-0.5

  v8f acc[6];
  #pragma unroll
  for (int t = 0; t < 6; ++t) acc[t] = zero8();

  v16h a0 = load_a16(X, r0, 0, DIM);
  v16h bf0[6];
  #pragma unroll
  for (int t = 0; t < 6; ++t) bf0[t] = load_b16(W, c0 + t * 16, 0, DIM);

  for (int kk = 0; kk < DIM; kk += 64) {
    // stage n+1
    v16h a1 = load_a16(X, r0, kk + 32, DIM);
    v16h bf1[6];
    #pragma unroll
    for (int t = 0; t < 6; ++t) bf1[t] = load_b16(W, c0 + t * 16, kk + 32, DIM);
    // compute n
    #pragma unroll
    for (int t = 0; t < 6; ++t) acc[t] = wmma16(a0, bf0[t], acc[t]);
    // stage n+2 (wraps harmlessly on last trip)
    const int kn = (kk + 64 < DIM) ? kk + 64 : 0;
    a0 = load_a16(X, r0, kn, DIM);
    #pragma unroll
    for (int t = 0; t < 6; ++t) bf0[t] = load_b16(W, c0 + t * 16, kn, DIM);
    // compute n+1
    #pragma unroll
    for (int t = 0; t < 6; ++t) acc[t] = wmma16(a1, bf1[t], acc[t]);
  }

  #pragma unroll
  for (int t = 0; t < 6; ++t) {
    const int c = c0 + t * 16 + ln;
    const float bval = bias[c];
    #pragma unroll
    for (int j = 0; j < 8; ++j) {
      const int r = r0 + j + 8 * hi;
      float val = acc[t][j] + bval;
      if (which == 0) {
        qp[(size_t)r * DIM + c] = (_Float16)(val * scale);
      } else if (which == 1) {
        kp[(size_t)r * DIM + c] = (_Float16)val;
      } else {
        const int bb = r >> 11;      // r / SEQ
        const int rn = r & 2047;     // r % SEQ
        vpT[((size_t)bb * DIM + c) * SEQ + rn] = (_Float16)val;
      }
    }
  }
}

// ---- kernel 2: cooperative flash attention ----------------------------------
// One 16-query tile per block, 4 waves. Keys processed 128 at a time:
//   - each wave computes its own 16x32 S strip (no redundant QK^T WMMAs)
//   - softmax stats merged across waves via LDS; P assembled once in LDS
//   - every wave consumes full P as A-fragments for its 192 output columns
// Q tile staged into LDS once via CDNA5 async-to-LDS DMA.
__global__ __launch_bounds__(128) void attn_kernel(
    const _Float16* __restrict__ qp, const _Float16* __restrict__ kp,
    const _Float16* __restrict__ vpT, _Float16* __restrict__ xbuf)
{
  __shared__ _Float16 qs[16 * DIM];      // 24 KB: Q tile (rows contiguous)
  __shared__ _Float16 pbuf[16 * 128];    //  4 KB: P tile (16 rows x 128 keys)
  __shared__ float    mbuf[4][16];       // per-wave strip row-max
  __shared__ float    lbuf[4][16];       // per-wave strip row-sum

  const int r0 = blockIdx.x * 16;
  const int bb = r0 >> 11;                 // batch index
  const int wave = threadIdx.x >> 5;
  const int lane = threadIdx.x & 31;
  const int hi = lane >> 4, ln = lane & 15;
  const int c0 = wave * 192;               // 4 waves * 192 = 768 columns

  const _Float16* kpb = kp  + (size_t)bb * SEQ * DIM;   // [2048, 768]
  const _Float16* vtb = vpT + (size_t)bb * DIM * SEQ;   // [768, 2048] (V^T)

  // ---- cooperative async DMA of the 16x768 Q tile into LDS (16 B / lane / op)
  {
    const _Float16* gsrc = qp + (size_t)r0 * DIM;   // tile is fully contiguous
    #pragma unroll
    for (int i = 0; i < 12; ++i) {
      const int e = ((int)threadIdx.x + i * 128) * 8;       // halves (16 B chunks)
      unsigned long long ga = (unsigned long long)(uintptr_t)(gsrc + e);
      unsigned int       la = (unsigned int)(uintptr_t)(&qs[e]);
      asm volatile("global_load_async_to_lds_b128 %0, %1, off"
                   :: "v"(la), "v"(ga) : "memory");
    }
    asm volatile("s_wait_asynccnt 0x0" ::: "memory");
    __syncthreads();
  }

  v8f acc[12];
  #pragma unroll
  for (int t = 0; t < 12; ++t) acc[t] = zero8();
  float mrow[8], lrow[8];
  #pragma unroll
  for (int j = 0; j < 8; ++j) { mrow[j] = -3.0e38f; lrow[j] = 0.f; }

  for (int it = 0; it < SEQ / 128; ++it) {
    const int j0 = it * 128;
    const int jw = j0 + wave * 32;          // this wave's 32-key strip

    // ---- S strip = Q @ K^T (16x32), double-buffered K fragments
    v8f s0 = zero8(), s1 = zero8();
    v16h kb0 = load_b16(kpb, jw,      0, DIM);
    v16h kb1 = load_b16(kpb, jw + 16, 0, DIM);
    for (int kk = 0; kk < DIM; kk += 32) {
      const int kn = (kk + 32 < DIM) ? kk + 32 : 0;
      v16h nb0 = load_b16(kpb, jw,      kn, DIM);
      v16h nb1 = load_b16(kpb, jw + 16, kn, DIM);
      v16h a = load_a16(qs, 0, kk, DIM);
      s0 = wmma16(a, kb0, s0);
      s1 = wmma16(a, kb1, s1);
      kb0 = nb0; kb1 = nb1;
    }

    // ---- strip row-max (xor reduce over the 16-lane group), publish
    float smax[8];
    #pragma unroll
    for (int j = 0; j < 8; ++j) {
      float mx = fmaxf(s0[j], s1[j]);
      #pragma unroll
      for (int off = 1; off < 16; off <<= 1)
        mx = fmaxf(mx, __shfl_xor(mx, off));
      smax[j] = mx;
    }
    if (ln == 0) {
      #pragma unroll
      for (int j = 0; j < 8; ++j) mbuf[wave][j + 8 * hi] = smax[j];
    }
    __syncthreads();   // B1: strip maxes visible

    // ---- combine maxes, rescale factors
    float alpha[8];
    #pragma unroll
    for (int j = 0; j < 8; ++j) {
      float g = mrow[j];
      #pragma unroll
      for (int w = 0; w < 4; ++w) g = fmaxf(g, mbuf[w][j + 8 * hi]);
      alpha[j] = __expf(mrow[j] - g);
      mrow[j] = g;
    }

    // ---- P strip = exp(S - m), strip row-sums, publish P + sums
    #pragma unroll
    for (int j = 0; j < 8; ++j) {
      const float p0 = __expf(s0[j] - mrow[j]);
      const float p1 = __expf(s1[j] - mrow[j]);
      s0[j] = p0; s1[j] = p1;
      float rs = p0 + p1;
      #pragma unroll
      for (int off = 1; off < 16; off <<= 1)
        rs += __shfl_xor(rs, off);
      smax[j] = rs;   // reuse as strip sum
    }
    if (ln == 0) {
      #pragma unroll
      for (int j = 0; j < 8; ++j) lbuf[wave][j + 8 * hi] = smax[j];
    }
    #pragma unroll
    for (int j = 0; j < 8; ++j) {
      const int m = j + 8 * hi;
      pbuf[m * 128 + wave * 32 + ln]      = (_Float16)s0[j];
      pbuf[m * 128 + wave * 32 + 16 + ln] = (_Float16)s1[j];
    }
    __syncthreads();   // B2: P tile + strip sums visible

    // ---- l update and accumulator rescale
    #pragma unroll
    for (int j = 0; j < 8; ++j) {
      float lsum = 0.f;
      #pragma unroll
      for (int w = 0; w < 4; ++w) lsum += lbuf[w][j + 8 * hi];
      lrow[j] = lrow[j] * alpha[j] + lsum;
    }
    #pragma unroll
    for (int t = 0; t < 12; ++t) {
      #pragma unroll
      for (int j = 0; j < 8; ++j) acc[t][j] *= alpha[j];
    }

    // ---- O += P @ V  (P A-frags from LDS, V B-frags contiguous from V^T)
    v16h pa[4];
    #pragma unroll
    for (int c = 0; c < 4; ++c) pa[c] = load_a16(pbuf, 0, c * 32, 128);
    #pragma unroll
    for (int t = 0; t < 12; ++t) {
      v16h bv0 = load_b16(vtb, c0 + t * 16, j0,      SEQ);
      v16h bv1 = load_b16(vtb, c0 + t * 16, j0 + 32, SEQ);
      v16h bv2 = load_b16(vtb, c0 + t * 16, j0 + 64, SEQ);
      v16h bv3 = load_b16(vtb, c0 + t * 16, j0 + 96, SEQ);
      acc[t] = wmma16(pa[0], bv0, acc[t]);
      acc[t] = wmma16(pa[1], bv1, acc[t]);
      acc[t] = wmma16(pa[2], bv2, acc[t]);
      acc[t] = wmma16(pa[3], bv3, acc[t]);
    }
  }

  // ---- normalize and store f16
  #pragma unroll
  for (int t = 0; t < 12; ++t) {
    #pragma unroll
    for (int j = 0; j < 8; ++j) {
      const int r = r0 + j + 8 * hi;
      const int c = c0 + t * 16 + ln;
      xbuf[(size_t)r * DIM + c] = (_Float16)(acc[t][j] / lrow[j]);
    }
  }
}

// ---- kernel 3: output projection (fp32 out), double-buffered ----------------
__global__ __launch_bounds__(256) void out_proj_kernel(
    const _Float16* __restrict__ xbuf, const _Float16* __restrict__ Wph,
    const float* __restrict__ bp, float* __restrict__ out)
{
  const int r0 = blockIdx.x * 16;
  const int wave = threadIdx.x >> 5;
  const int lane = threadIdx.x & 31;
  const int hi = lane >> 4, ln = lane & 15;
  const int c0 = wave * 96;

  v8f acc[6];
  #pragma unroll
  for (int t = 0; t < 6; ++t) acc[t] = zero8();

  v16h a0 = load_a16(xbuf, r0, 0, DIM);
  v16h bf0[6];
  #pragma unroll
  for (int t = 0; t < 6; ++t) bf0[t] = load_b16(Wph, c0 + t * 16, 0, DIM);

  for (int kk = 0; kk < DIM; kk += 64) {
    v16h a1 = load_a16(xbuf, r0, kk + 32, DIM);
    v16h bf1[6];
    #pragma unroll
    for (int t = 0; t < 6; ++t) bf1[t] = load_b16(Wph, c0 + t * 16, kk + 32, DIM);
    #pragma unroll
    for (int t = 0; t < 6; ++t) acc[t] = wmma16(a0, bf0[t], acc[t]);
    const int kn = (kk + 64 < DIM) ? kk + 64 : 0;
    a0 = load_a16(xbuf, r0, kn, DIM);
    #pragma unroll
    for (int t = 0; t < 6; ++t) bf0[t] = load_b16(Wph, c0 + t * 16, kn, DIM);
    #pragma unroll
    for (int t = 0; t < 6; ++t) acc[t] = wmma16(a1, bf1[t], acc[t]);
  }

  #pragma unroll
  for (int t = 0; t < 6; ++t) {
    const int c = c0 + t * 16 + ln;
    const float bval = bp[c];
    #pragma unroll
    for (int j = 0; j < 8; ++j) {
      out[(size_t)(r0 + j + 8 * hi) * DIM + c] = acc[t][j] + bval;
    }
  }
}

// ---- host launch ------------------------------------------------------------
extern "C" void kernel_launch(void* const* d_in, const int* in_sizes, int n_in,
                              void* d_out, int out_size, void* d_ws, size_t ws_size,
                              hipStream_t stream) {
  const float* q  = (const float*)d_in[0];
  const float* k  = (const float*)d_in[1];
  const float* v  = (const float*)d_in[2];
  const float* Wq = (const float*)d_in[3];
  const float* bq = (const float*)d_in[4];
  const float* Wk = (const float*)d_in[5];
  const float* bk = (const float*)d_in[6];
  const float* Wv = (const float*)d_in[7];
  const float* bv = (const float*)d_in[8];
  const float* Wp = (const float*)d_in[9];
  const float* bp = (const float*)d_in[10];
  float* out = (float*)d_out;

  // workspace layout (f16):
  //   qh|kh|vh|qp|kp|vpT|xbuf : NTOK*DIM each;  Wqh|Wkh|Wvh|Wph : DIM*DIM each
  _Float16* wsh = (_Float16*)d_ws;
  const size_t SZ  = (size_t)NTOK * DIM;   // 6,291,456
  const size_t WSZ = (size_t)DIM * DIM;    //   589,824
  _Float16* qh   = wsh;
  _Float16* kh   = wsh + 1 * SZ;
  _Float16* vh   = wsh + 2 * SZ;
  _Float16* qp   = wsh + 3 * SZ;
  _Float16* kp   = wsh + 4 * SZ;
  _Float16* vpT  = wsh + 5 * SZ;
  _Float16* xbuf = wsh + 6 * SZ;
  _Float16* Wqh  = wsh + 7 * SZ;
  _Float16* Wkh  = Wqh + WSZ;
  _Float16* Wvh  = Wqh + 2 * WSZ;
  _Float16* Wph  = Wqh + 3 * WSZ;

  const int nb_big = (int)(SZ  / (256 * 8));   // 3072 blocks
  const int nb_w   = (int)(WSZ / (256 * 8));   //  288 blocks
  cvt_kernel<<<nb_big, 256, 0, stream>>>(q,  qh,  (int)SZ);
  cvt_kernel<<<nb_big, 256, 0, stream>>>(k,  kh,  (int)SZ);
  cvt_kernel<<<nb_big, 256, 0, stream>>>(v,  vh,  (int)SZ);
  cvt_kernel<<<nb_w,   256, 0, stream>>>(Wq, Wqh, (int)WSZ);
  cvt_kernel<<<nb_w,   256, 0, stream>>>(Wk, Wkh, (int)WSZ);
  cvt_kernel<<<nb_w,   256, 0, stream>>>(Wv, Wvh, (int)WSZ);
  cvt_kernel<<<nb_w,   256, 0, stream>>>(Wp, Wph, (int)WSZ);

  qkv_proj_kernel<<<dim3(NTOK / 16, 3), 256, 0, stream>>>(
      qh, kh, vh, Wqh, Wkh, Wvh, bq, bk, bv, qp, kp, vpT);
  attn_kernel<<<NTOK / 16, 128, 0, stream>>>(qp, kp, vpT, xbuf);
  out_proj_kernel<<<NTOK / 16, 256, 0, stream>>>(xbuf, Wph, bp, out);
}
